// GRUnet_75376676045371
// MI455X (gfx1250) — compile-verified, matrix operand order checked by reading, not run on hardware
//
#include <hip/hip_runtime.h>

// ---------------------------------------------------------------------------
// Types for CDNA5 WMMA (wave32): v_wmma_f32_16x16x32_bf16
// ---------------------------------------------------------------------------
typedef __attribute__((ext_vector_type(16))) __bf16 v16bf;
typedef __attribute__((ext_vector_type(8)))  float  v8f;
typedef __attribute__((ext_vector_type(4)))  unsigned int u32x4;
typedef __attribute__((ext_vector_type(8)))  int i32x8;
typedef __attribute__((ext_vector_type(4)))  int i32x4;

union Frag16 {
    uint4 q[2];
    v16bf v;
};
static_assert(sizeof(Frag16) == 32, "frag size");

__device__ __forceinline__ unsigned short f2bf(float f) {
    unsigned int u = __float_as_uint(f);
    unsigned int r = u + 0x7FFFu + ((u >> 16) & 1u);   // round-to-nearest-even
    return (unsigned short)(r >> 16);
}

__device__ __forceinline__ float sigmoidf_fast(float x) {
    return 1.0f / (1.0f + __expf(-x));
}

// Low 32 bits of a flat LDS pointer == LDS byte offset.
__device__ __forceinline__ unsigned lds_offset(const void* p) {
    return (unsigned)(size_t)p;
}

// Async 16B global->LDS copy (per-lane), tracked by ASYNCcnt.
__device__ __forceinline__ void async_copy_b128(unsigned lds_off, const void* gsrc) {
    asm volatile("global_load_async_to_lds_b128 %0, %1, off"
                 :: "v"(lds_off), "v"(gsrc) : "memory");
}
__device__ __forceinline__ void wait_asynccnt0() {
    asm volatile("s_wait_asynccnt 0x0" ::: "memory");
}
__device__ __forceinline__ void wait_asynccnt4() {
    asm volatile("s_wait_asynccnt 0x4" ::: "memory");
}

// ---------------------------------------------------------------------------
// Problem constants
// ---------------------------------------------------------------------------
#define H_IN   1342
#define H_INP  1344        // K padded to multiple of 32
#define FC_NP  1408        // FC N padded to multiple of 128
#define HC     256
#define G3     768         // 3*HC
#define BB     128
#define TT     128
#define NWG_REC 8

// ---------------------------------------------------------------------------
// fp32 -> bf16 converters (with zero padding)
// ---------------------------------------------------------------------------
__global__ void k_cvt_x(const float* __restrict__ X, unsigned short* __restrict__ Xb) {
    // X: (B, T, H_IN) fp32  ->  Xb: (T*B, H_INP) bf16, row = t*B + b
    size_t idx = (size_t)blockIdx.x * blockDim.x + threadIdx.x;
    size_t total = (size_t)TT * BB * H_INP;
    if (idx >= total) return;
    int k = (int)(idx % H_INP);
    size_t row = idx / H_INP;
    int t = (int)(row / BB);
    int b = (int)(row % BB);
    float v = (k < H_IN) ? X[((size_t)b * TT + t) * H_IN + k] : 0.0f;
    Xb[idx] = f2bf(v);
}

__global__ void k_cvt_w(const float* __restrict__ W, unsigned short* __restrict__ Wb,
                        int N, int K, int Np, int Kp) {
    size_t idx = (size_t)blockIdx.x * blockDim.x + threadIdx.x;
    if (idx >= (size_t)Np * Kp) return;
    int k = (int)(idx % Kp);
    int n = (int)(idx / Kp);
    float v = (n < N && k < K) ? W[(size_t)n * K + k] : 0.0f;
    Wb[idx] = f2bf(v);
}

__global__ void k_zero_u32(unsigned int* __restrict__ p, int n) {
    int i = blockIdx.x * blockDim.x + threadIdx.x;
    if (i < n) p[i] = 0u;
}

// ---------------------------------------------------------------------------
// LDS-tiled bf16 WMMA GEMM:  C[m,n] = sum_k A[m,k]*W[n,k] + bias[n] (fp32 out)
// WG = 256 threads (8 waves); output tile 128(M) x 128(N); K staged in 32-wide
// triple-buffered LDS slabs filled with global_load_async_to_lds_b128 (4 b128
// copies per thread per slab, branch-free; async loads complete in order ->
// s_wait_asynccnt 4 retires exactly the oldest slab).  Last two K iterations
// are peeled so the steady-state loop has no conditional wait selection.
// Wave w: rows [32*(w&3), +32), cols [64*(w>>2), +64): 2x4 WMMA accumulators,
// 4 B-frags hoisted to registers -> 12 ds_load_b128 per 8 WMMAs.
// LDS rows padded to 40 elements so b128 fragment reads are conflict-free.
// ---------------------------------------------------------------------------
#define GMT 128
#define GNT 128
#define AST 40
#define NBUF 3

__global__ void k_gemm_lds(const unsigned short* __restrict__ A, int lda,
                           const unsigned short* __restrict__ W, int ldb,
                           const float* __restrict__ bias,
                           float* __restrict__ C, int ldc,
                           int M, int Npad, int Nact, int K) {
    __shared__ __align__(16) unsigned short ldsA[NBUF][GMT * AST];
    __shared__ __align__(16) unsigned short ldsB[NBUF][GNT * AST];

    const int ntiles = Npad / GNT;
    const int m_base = (blockIdx.x / ntiles) * GMT;
    const int n_base = (blockIdx.x % ntiles) * GNT;

    const int tid  = threadIdx.x;
    const int lane = tid & 31;
    const int wave = tid >> 5;
    const int sel  = lane >> 4;
    const int ln   = lane & 15;
    const int wm   = wave & 3;      // 32-row block
    const int wn   = wave >> 2;     // 64-col block

    const int nslab = K >> 5;

    // Branch-free per-thread staging: 2 A rows + 2 B rows, 16B chunk each.
    const int srow = tid >> 2;      // 0..63
    const int part = tid & 3;
    const unsigned short* gA0 = A + (size_t)(m_base + srow)      * lda + part * 8;
    const unsigned short* gA1 = A + (size_t)(m_base + 64 + srow) * lda + part * 8;
    const unsigned short* gB0 = W + (size_t)(n_base + srow)      * ldb + part * 8;
    const unsigned short* gB1 = W + (size_t)(n_base + 64 + srow) * ldb + part * 8;
    const unsigned dA0 = lds_offset(&ldsA[0][srow * AST + part * 8]);
    const unsigned dA1 = lds_offset(&ldsA[0][(64 + srow) * AST + part * 8]);
    const unsigned dB0 = lds_offset(&ldsB[0][srow * AST + part * 8]);
    const unsigned dB1 = lds_offset(&ldsB[0][(64 + srow) * AST + part * 8]);
    const unsigned bufA = (unsigned)(GMT * AST * 2);
    const unsigned bufB = (unsigned)(GNT * AST * 2);

    auto stage = [&](int s, int p) {
        const int k0 = s << 5;
        async_copy_b128(dA0 + (unsigned)p * bufA, gA0 + k0);
        async_copy_b128(dA1 + (unsigned)p * bufA, gA1 + k0);
        async_copy_b128(dB0 + (unsigned)p * bufB, gB0 + k0);
        async_copy_b128(dB1 + (unsigned)p * bufB, gB1 + k0);
    };

    v8f acc[2][4];
    #pragma unroll
    for (int mi = 0; mi < 2; ++mi)
        #pragma unroll
        for (int ni = 0; ni < 4; ++ni)
            #pragma unroll
            for (int i = 0; i < 8; ++i) acc[mi][ni][i] = 0.0f;

    auto compute = [&](int p) {
        Frag16 fbv[4];
        #pragma unroll
        for (int ni = 0; ni < 4; ++ni) {
            const int br = wn * 64 + ni * 16 + ln;
            fbv[ni].q[0] = *(const uint4*)&ldsB[p][br * AST + 8 * sel];
            fbv[ni].q[1] = *(const uint4*)&ldsB[p][br * AST + 16 + 8 * sel];
        }
        #pragma unroll
        for (int mi = 0; mi < 2; ++mi) {
            Frag16 fa;
            const int ar = wm * 32 + mi * 16 + ln;
            fa.q[0] = *(const uint4*)&ldsA[p][ar * AST + 8 * sel];
            fa.q[1] = *(const uint4*)&ldsA[p][ar * AST + 16 + 8 * sel];
            #pragma unroll
            for (int ni = 0; ni < 4; ++ni) {
                acc[mi][ni] = __builtin_amdgcn_wmma_f32_16x16x32_bf16(
                    false, fa.v, false, fbv[ni].v, (short)0, acc[mi][ni], false, false);
            }
        }
    };

    stage(0, 0);
    if (nslab > 1) { stage(1, 1); wait_asynccnt4(); }
    else           { wait_asynccnt0(); }
    __syncthreads();

    int p = 0;
    // Steady state: unconditional stage -> compute -> wait4 -> barrier.
    for (int s = 0; s + 2 < nslab; ++s) {
        const int q = (p + 2 >= NBUF) ? p + 2 - NBUF : p + 2;
        stage(s + 2, q);              // overlaps this slab's compute
        compute(p);
        wait_asynccnt4();             // oldest in-flight slab retired
        __syncthreads();
        p = (p + 1 == NBUF) ? 0 : p + 1;
    }
    // Peeled slab nslab-2: no stage; drain remaining async before last slab.
    if (nslab >= 2) {
        compute(p);
        wait_asynccnt0();
        __syncthreads();
        p = (p + 1 == NBUF) ? 0 : p + 1;
    }
    // Peeled final slab: no stage, no wait.
    compute(p);

    #pragma unroll
    for (int mi = 0; mi < 2; ++mi)
        #pragma unroll
        for (int ni = 0; ni < 4; ++ni) {
            int col = n_base + wn * 64 + ni * 16 + ln;
            if (col < Nact) {
                float bv = bias ? bias[col] : 0.0f;
                #pragma unroll
                for (int i = 0; i < 8; ++i) {
                    int row = m_base + wm * 32 + mi * 16 + i + 8 * sel;
                    C[(size_t)row * ldc + col] = acc[mi][ni][i] + bv;
                }
            }
        }
}

// ---------------------------------------------------------------------------
// Persistent GRU layer recurrence (8 WGs x 8 waves).
// WG wg owns h columns [32*wg, +32). Its W_hh slice (3 gates x 32 rows x 256)
// is DMA'd into LDS by the Tensor Data Mover: 3 2D-tile descriptors whose
// pad feature (128 DWORDs -> +4 DWORDs) produces the 264-element padded rows
// needed for conflict-free b128 B-fragment reads.  h state is double-buffered
// in global memory (L2-resident) behind a monotonic agent-scope spin barrier.
// ---------------------------------------------------------------------------
#define LDS_STRIDE 264

__device__ __forceinline__ void tdm_load_tile_2d(unsigned lds_byte_off,
                                                 const void* gsrc,
                                                 int rows /*tile_dim1*/,
                                                 int rowlen /*tile_dim0 elems*/) {
    unsigned long long ga = (unsigned long long)(size_t)gsrc;
    u32x4 g0;
    g0.x = 1u;                                             // count=1, user descriptor
    g0.y = lds_byte_off;                                   // lds_addr
    g0.z = (unsigned)(ga & 0xFFFFFFFFull);                 // global_addr[31:0]
    g0.w = (unsigned)((ga >> 32) & 0x1FFFFFFull)           // global_addr[56:32]
         | (2u << 30);                                     // type = 2 ("image")
    i32x8 g1;
    g1[0] = (int)((1u << 16)      // data_size = 1 -> 2 bytes
                | (1u << 20)      // pad_enable
                | (6u << 22)      // pad_interval code 6 = 128 DWORDs (one 512B row)
                | (3u << 25));    // pad_amount  code 3 = 4 DWORDs (16B)
    g1[1] = (int)((unsigned)rowlen << 16);                 // tensor_dim0[15:0]
    g1[2] = (int)((unsigned)rows << 16);                   // tensor_dim1[15:0]
    g1[3] = (int)((unsigned)rowlen << 16);                 // tile_dim0
    g1[4] = rows;                                          // tile_dim1 (tile_dim2=0)
    g1[5] = rowlen;                                        // tensor_dim0_stride[31:0]
    g1[6] = 0;
    g1[7] = 0;
    i32x4 gz = {0, 0, 0, 0};
#if __clang_major__ >= 23
    i32x8 gz8 = {0, 0, 0, 0, 0, 0, 0, 0};
    __builtin_amdgcn_tensor_load_to_lds(g0, g1, gz, gz, gz8, 0);
#else
    __builtin_amdgcn_tensor_load_to_lds(g0, g1, gz, gz, 0);
#endif
}

__global__ void k_gru_layer(const float* __restrict__ gx,            // (T*B, 768)
                            const unsigned short* __restrict__ Whh,  // (768, 256) bf16
                            const float* __restrict__ bhh,           // (768)
                            const float* __restrict__ h0src,         // (B, 256) fp32
                            float* __restrict__ h32,                 // 2 x (B,256)
                            unsigned short* __restrict__ hb16,       // 2 x (B,256)
                            unsigned short* __restrict__ ys,         // (T*B, 256) bf16
                            unsigned short* __restrict__ hcat,       // (B, 768) bf16
                            int layer,
                            unsigned int* __restrict__ bar) {
    __shared__ __align__(16) unsigned short ldsW[96 * LDS_STRIDE];

    const int wg   = blockIdx.x;          // 0..7
    const int lane = threadIdx.x & 31;
    const int wave = threadIdx.x >> 5;    // 0..7 -> M tile
    const int sel  = lane >> 4;
    const int ln   = lane & 15;
    const int col_base = wg * 32;

    // TDM: stage the 3 gate slices of W_hh (32 rows x 256 each) into LDS.
    if (wave == 0) {
        #pragma unroll
        for (int gate = 0; gate < 3; ++gate) {
            tdm_load_tile_2d(lds_offset(&ldsW[0]) + (unsigned)(gate * 32 * LDS_STRIDE * 2),
                             Whh + (size_t)(gate * 256 + col_base) * HC,
                             32, HC);
        }
        __builtin_amdgcn_s_wait_tensorcnt(0);
    }

    // Init h (buffer 0) for the owned column slice.
    for (int c = threadIdx.x; c < BB * 32; c += blockDim.x) {
        int row = c >> 5;
        int col = col_base + (c & 31);
        float v = h0src[row * HC + col];
        h32[row * HC + col] = v;
        hb16[row * HC + col] = f2bf(v);
    }
    __threadfence();
    __syncthreads();
    if (threadIdx.x == 0) {
        __hip_atomic_fetch_add(bar, 1u, __ATOMIC_ACQ_REL, __HIP_MEMORY_SCOPE_AGENT);
        while (__hip_atomic_load(bar, __ATOMIC_ACQUIRE, __HIP_MEMORY_SCOPE_AGENT)
               < (unsigned)NWG_REC) { }
    }
    __syncthreads();

    const int m0 = wave * 16;
    float bh[3][2];
    #pragma unroll
    for (int g = 0; g < 3; ++g)
        #pragma unroll
        for (int cg = 0; cg < 2; ++cg)
            bh[g][cg] = bhh[g * 256 + col_base + cg * 16 + ln];

    for (int t = 0; t < TT; ++t) {
        const int rb = t & 1;
        const int wb = rb ^ 1;
        const unsigned short* hb_r = hb16 + (size_t)rb * BB * HC;
        const float*          h32r = h32  + (size_t)rb * BB * HC;
        float*                h32w = h32  + (size_t)wb * BB * HC;
        unsigned short*       hb_w = hb16 + (size_t)wb * BB * HC;

        v8f acc[3][2];
        #pragma unroll
        for (int g = 0; g < 3; ++g)
            #pragma unroll
            for (int cg = 0; cg < 2; ++cg)
                #pragma unroll
                for (int i = 0; i < 8; ++i) acc[g][cg][i] = bh[g][cg];

        const unsigned short* arow = hb_r + (size_t)(m0 + ln) * HC;
        for (int k0 = 0; k0 < HC; k0 += 32) {
            v16bf a;
            {
                Frag16 fa;
                fa.q[0] = *(const uint4*)(arow + k0 + 8 * sel);
                fa.q[1] = *(const uint4*)(arow + k0 + 16 + 8 * sel);
                a = fa.v;
            }
            #pragma unroll
            for (int g = 0; g < 3; ++g) {
                #pragma unroll
                for (int cg = 0; cg < 2; ++cg) {
                    int rl = g * 32 + cg * 16 + ln;
                    Frag16 fb;
                    fb.q[0] = *(const uint4*)&ldsW[rl * LDS_STRIDE + k0 + 8 * sel];
                    fb.q[1] = *(const uint4*)&ldsW[rl * LDS_STRIDE + k0 + 16 + 8 * sel];
                    acc[g][cg] = __builtin_amdgcn_wmma_f32_16x16x32_bf16(
                        false, a, false, fb.v, (short)0, acc[g][cg], false, false);
                }
            }
        }

        const float* gxt = gx + (size_t)t * BB * G3;
        #pragma unroll
        for (int cg = 0; cg < 2; ++cg) {
            int col = col_base + cg * 16 + ln;
            #pragma unroll
            for (int i = 0; i < 8; ++i) {
                int row = m0 + i + 8 * sel;
                const float* gxr = gxt + (size_t)row * G3;
                float r = sigmoidf_fast(gxr[col]       + acc[0][cg][i]);
                float z = sigmoidf_fast(gxr[256 + col] + acc[1][cg][i]);
                float n = tanhf(gxr[512 + col] + r * acc[2][cg][i]);
                float hp = h32r[row * HC + col];
                float hn = (1.0f - z) * n + z * hp;
                unsigned short hb = f2bf(hn);
                h32w[row * HC + col] = hn;
                hb_w[row * HC + col] = hb;
                ys[((size_t)t * BB + row) * HC + col] = hb;
                if (t == TT - 1)
                    hcat[(size_t)row * G3 + layer * HC + col] = hb;
            }
        }

        if (t != TT - 1) {
            __threadfence();
            __syncthreads();
            if (threadIdx.x == 0) {
                __hip_atomic_fetch_add(bar, 1u, __ATOMIC_ACQ_REL, __HIP_MEMORY_SCOPE_AGENT);
                unsigned target = (unsigned)NWG_REC * (t + 2);
                while (__hip_atomic_load(bar, __ATOMIC_ACQUIRE, __HIP_MEMORY_SCOPE_AGENT)
                       < target) { }
            }
            __syncthreads();
        }
    }
}

// ---------------------------------------------------------------------------
// Host side
// ---------------------------------------------------------------------------
extern "C" void kernel_launch(void* const* d_in, const int* in_sizes, int n_in,
                              void* d_out, int out_size, void* d_ws, size_t ws_size,
                              hipStream_t stream) {
    (void)in_sizes; (void)n_in; (void)out_size; (void)ws_size;

    const float* X      = (const float*)d_in[0];
    const float* hidden = (const float*)d_in[1];
    const float* w_ih[3] = {(const float*)d_in[2], (const float*)d_in[6],  (const float*)d_in[10]};
    const float* w_hh[3] = {(const float*)d_in[3], (const float*)d_in[7],  (const float*)d_in[11]};
    const float* b_ih[3] = {(const float*)d_in[4], (const float*)d_in[8],  (const float*)d_in[12]};
    const float* b_hh[3] = {(const float*)d_in[5], (const float*)d_in[9],  (const float*)d_in[13]};
    const float* fc_w = (const float*)d_in[14];
    const float* fc_b = (const float*)d_in[15];

    char* ws = (char*)d_ws;
    const size_t o_Xb   = 0;                                   // 16384*1344*2
    const size_t o_Wih0 = o_Xb   + (size_t)TT*BB*H_INP*2;      // 768*1344*2
    const size_t o_Whh0 = o_Wih0 + (size_t)G3*H_INP*2;
    const size_t o_Whh1 = o_Whh0 + (size_t)G3*HC*2;
    const size_t o_Whh2 = o_Whh1 + (size_t)G3*HC*2;
    const size_t o_Wih1 = o_Whh2 + (size_t)G3*HC*2;
    const size_t o_Wih2 = o_Wih1 + (size_t)G3*HC*2;
    const size_t o_fcW  = o_Wih2 + (size_t)G3*HC*2;            // 1408*768*2
    const size_t o_gx   = o_fcW  + (size_t)FC_NP*G3*2;         // 16384*768*4
    const size_t o_ys   = o_gx   + (size_t)TT*BB*G3*4;         // 16384*256*2
    const size_t o_h32  = o_ys   + (size_t)TT*BB*HC*2;         // 2*128*256*4
    const size_t o_hb16 = o_h32  + (size_t)2*BB*HC*4;          // 2*128*256*2
    const size_t o_hcat = o_hb16 + (size_t)2*BB*HC*2;          // 128*768*2
    const size_t o_bar  = o_hcat + (size_t)BB*G3*2;

    unsigned short* Xb    = (unsigned short*)(ws + o_Xb);
    unsigned short* Wih0b = (unsigned short*)(ws + o_Wih0);
    unsigned short* Whhb[3] = {(unsigned short*)(ws + o_Whh0),
                               (unsigned short*)(ws + o_Whh1),
                               (unsigned short*)(ws + o_Whh2)};
    unsigned short* Wihb12[2] = {(unsigned short*)(ws + o_Wih1),
                                 (unsigned short*)(ws + o_Wih2)};
    unsigned short* fcWb  = (unsigned short*)(ws + o_fcW);
    float*          gx    = (float*)(ws + o_gx);
    unsigned short* ys    = (unsigned short*)(ws + o_ys);
    float*          h32   = (float*)(ws + o_h32);
    unsigned short* hb16  = (unsigned short*)(ws + o_hb16);
    unsigned short* hcat  = (unsigned short*)(ws + o_hcat);
    unsigned int*   bar   = (unsigned int*)(ws + o_bar);

    {
        size_t tot = (size_t)TT * BB * H_INP;
        k_cvt_x<<<dim3((unsigned)((tot + 255) / 256)), dim3(256), 0, stream>>>(X, Xb);
    }
    auto cvt = [&](const float* src, unsigned short* dst, int N, int K, int Np, int Kp) {
        size_t tot = (size_t)Np * Kp;
        k_cvt_w<<<dim3((unsigned)((tot + 255) / 256)), dim3(256), 0, stream>>>(src, dst, N, K, Np, Kp);
    };
    cvt(w_ih[0], Wih0b, G3, H_IN, G3, H_INP);
    cvt(w_hh[0], Whhb[0], G3, HC, G3, HC);
    cvt(w_hh[1], Whhb[1], G3, HC, G3, HC);
    cvt(w_hh[2], Whhb[2], G3, HC, G3, HC);
    cvt(w_ih[1], Wihb12[0], G3, HC, G3, HC);
    cvt(w_ih[2], Wihb12[1], G3, HC, G3, HC);
    cvt(fc_w, fcWb, H_IN, G3, FC_NP, G3);

    k_zero_u32<<<dim3(1), dim3(64), 0, stream>>>(bar, 16);

    const int M = TT * BB;                  // 16384
    auto gemm = [&](const unsigned short* A, int lda, const unsigned short* W, int ldb,
                    const float* bias, float* C, int ldc,
                    int Mm, int Npad, int Nact, int K) {
        int blocks = (Mm / GMT) * (Npad / GNT);
        k_gemm_lds<<<dim3(blocks), dim3(256), 0, stream>>>(
            A, lda, W, ldb, bias, C, ldc, Mm, Npad, Nact, K);
    };

    // ---- layer 0 -----------------------------------------------------------
    gemm(Xb, H_INP, Wih0b, H_INP, b_ih[0], gx, G3, M, G3, G3, H_INP);
    k_gru_layer<<<dim3(NWG_REC), dim3(256), 0, stream>>>(
        gx, Whhb[0], b_hh[0], hidden + 0 * BB * HC,
        h32, hb16, ys, hcat, 0, bar + 0);

    // ---- layer 1 -----------------------------------------------------------
    gemm(ys, HC, Wihb12[0], HC, b_ih[1], gx, G3, M, G3, G3, HC);
    k_gru_layer<<<dim3(NWG_REC), dim3(256), 0, stream>>>(
        gx, Whhb[1], b_hh[1], hidden + 1 * BB * HC,
        h32, hb16, ys, hcat, 1, bar + 1);

    // ---- layer 2 -----------------------------------------------------------
    gemm(ys, HC, Wihb12[1], HC, b_ih[2], gx, G3, M, G3, G3, HC);
    k_gru_layer<<<dim3(NWG_REC), dim3(256), 0, stream>>>(
        gx, Whhb[2], b_hh[2], hidden + 2 * BB * HC,
        h32, hb16, ys, hcat, 2, bar + 2);

    // ---- FC: out[b, j] = hcat[b, :] . fc_w[j, :] + fc_b[j] ----------------
    gemm(hcat, G3, fcWb, G3, fc_b, (float*)d_out, H_IN, BB, FC_NP, H_IN, G3);

    // ---- second tuple output: hidden passed through unchanged -------------
    hipMemcpyAsync((float*)d_out + (size_t)BB * H_IN, hidden,
                   (size_t)3 * BB * HC * sizeof(float),
                   hipMemcpyDeviceToDevice, stream);
}
